// TreeLSTM_39479339385453
// MI455X (gfx1250) — matrix operand order repaired and checked by reading, not compiled, not run on hardware
//
#include <hip/hip_runtime.h>
#include <hip/hip_bf16.h>

// ---------------- CDNA5 WMMA types ----------------
typedef __bf16 bf16_t;
typedef bf16_t v16bf __attribute__((ext_vector_type(16)));
typedef bf16_t v8bf  __attribute__((ext_vector_type(8)));
typedef bf16_t v4bf  __attribute__((ext_vector_type(4)));
typedef float  v8f   __attribute__((ext_vector_type(8)));

#define FEAT 256
#define HID  256

// branch-free fast sigmoid/tanh: v_exp_f32 + v_rcp_f32, no libm range-split divergence
static __device__ __forceinline__ float sigm(float x) {
  return __builtin_amdgcn_rcpf(1.0f + __expf(-x));
}
static __device__ __forceinline__ float fast_tanh(float x) {
  float e = __expf(-2.0f * __builtin_fabsf(x));
  float t = (1.0f - e) * __builtin_amdgcn_rcpf(1.0f + e);
  return __builtin_copysignf(t, x);
}

// A-fragment from fragment-swizzled LDS: [pt][k][lane][16] -> one aligned 32B read/lane,
// consecutive lanes hit consecutive 32B chunks => conflict-free, no padding needed.
static __device__ __forceinline__ v16bf load_a(const bf16_t* __restrict__ s, int pt, int k, int lane) {
  return *(const v16bf*)(s + ((((((pt << 3) + k) << 5) + lane) << 4)));
}

// B-fragment (32x16 bf16) from pre-packed weights: one contiguous 32B load per lane
static __device__ __forceinline__ v16bf load_b(const bf16_t* __restrict__ Bp, int ntiles, int nt,
                                               int k, int lane) {
  return *(const v16bf*)(Bp + (((((long long)k * ntiles + nt) << 5) + lane) << 4));
}

// packed bf16 add (v_pk_add_bf16 co-executes with XDL WMMA)
static __device__ __forceinline__ v16bf add_bf(v16bf a, v16bf b) { return a + b; }

#define WMMA_BF16(A, B, C) \
  __builtin_amdgcn_wmma_f32_16x16x32_bf16(false, (A), false, (B), (short)0, (C), false, false)

// -------- pack row-major f32 weights (K x Ncols) into B-fragment-linear bf16 --------
__global__ __launch_bounds__(256) void pack_weights_kernel(const float* __restrict__ W,
                                                           bf16_t* __restrict__ Wp, int K, int Ncols) {
  int idx = blockIdx.x * blockDim.x + threadIdx.x;
  if (idx >= K * Ncols) return;
  int j      = idx & 15;
  int lane   = (idx >> 4) & 31;
  int tile   = idx >> 9;
  int ntiles = Ncols >> 4;
  int nt     = tile % ntiles;
  int kt     = tile / ntiles;
  int krow   = (kt << 5) + ((j >> 3) << 4) + ((lane >> 4) << 3) + (j & 7);  // lanes 0-15: K=0..7,16..23 of block; 16-31: K=8..15,24..31
  int col    = (nt << 4) + (lane & 15);
  Wp[idx] = (bf16_t)W[(long long)krow * Ncols + col];
}

// -------- one tree level: 32 parents per workgroup (2 sub-tiles), 8 waves x 2 hid-tiles --------
// Every B fragment is reused by both parent sub-tiles => half the L2 weight traffic.
// k-loop deliberately kept rolled: bounds per-iteration register liveness so multiple
// waves fit per SIMD and hide the L2 B-fragment latency via occupancy.
template <bool HC>
__global__ __launch_bounds__(256) void level_kernel(
    const float*  __restrict__ x,      // N x 256 features (f32)
    bf16_t*       __restrict__ h,      // N x 256 (bf16)
    float*        __restrict__ c,      // N x 256 (f32)
    const bf16_t* __restrict__ Wip,    // packed 256x768
    const bf16_t* __restrict__ Uip,    // packed 256x768
    const bf16_t* __restrict__ Wfp,    // packed 256x256
    const bf16_t* __restrict__ Ufp,    // packed 256x256
    const float*  __restrict__ b_iou,  // 768
    const float*  __restrict__ b_f,    // 256
    int node_base, int child_base) {
  // fragment-swizzled staging buffers: [pt:2][k:8][lane:32][j:16] bf16 = 16KB each (48KB total)
  __shared__ __attribute__((aligned(32))) bf16_t sX [2 * 8 * 32 * 16];
  __shared__ __attribute__((aligned(32))) bf16_t sH0[2 * 8 * 32 * 16];
  __shared__ __attribute__((aligned(32))) bf16_t sH1[2 * 8 * 32 * 16];

  const int tid  = threadIdx.x;
  const int tile = blockIdx.x;
  const int p0   = node_base + (tile << 5);

  // ---- stage features: float4 reads, swizzle into fragment layout ----
#pragma unroll
  for (int e = tid; e < 32 * 64; e += 256) {  // (row, 4-col group)
    int r = e >> 6;             // 0..31
    int q = (e & 63) << 2;      // col, step 4
    float4 xv = *(const float4*)(x + (long long)(p0 + r) * FEAT + q);
    int k        = q >> 5;
    int w        = q & 31;
    int lanehalf = (w >> 3) & 1;
    int j        = (w & 7) + ((w >> 4) << 3);
    int lane     = (r & 15) | (lanehalf << 4);
    int pt       = r >> 4;
    v4bf xb;
    xb[0] = (bf16_t)xv.x; xb[1] = (bf16_t)xv.y; xb[2] = (bf16_t)xv.z; xb[3] = (bf16_t)xv.w;
    *(v4bf*)(sX + (((((pt << 3) + k) << 5) + lane) << 4) + j) = xb;
  }
  if (HC) {
    // ---- stage children h (bf16, v8bf reads); h_sum fragments are formed in-register later ----
#pragma unroll
    for (int e = tid; e < 32 * 32; e += 256) {  // (row, 8-col group)
      int r = e >> 5;            // 0..31
      int q = (e & 31) << 3;     // col, step 8
      long long ch0 = child_base + (long long)(((tile << 5) + r)) * 2;
      v8bf a0 = *(const v8bf*)(h + ch0 * HID + q);
      v8bf a1 = *(const v8bf*)(h + (ch0 + 1) * HID + q);
      int k        = q >> 5;
      int w        = q & 31;           // 0,8,16,24
      int lanehalf = (w >> 3) & 1;
      int j        = (w >> 4) << 3;    // 0 or 8
      int lane     = (r & 15) | (lanehalf << 4);
      int pt       = r >> 4;
      int base     = ((((pt << 3) + k) << 5) + lane) << 4;
      *(v8bf*)(sH0 + base + j) = a0;
      *(v8bf*)(sH1 + base + j) = a1;
    }
  }
  __syncthreads();

  const int wave = tid >> 5;
  const int lane = tid & 31;

#pragma unroll 1
  for (int ti = 0; ti < 2; ++ti) {
    const int t = (wave << 1) | ti;  // hid column tile 0..15
    v8f zero = {0.f, 0.f, 0.f, 0.f, 0.f, 0.f, 0.f, 0.f};
    v8f ai[2]  = {zero, zero}, ao[2] = {zero, zero}, au[2] = {zero, zero};
    v8f afx[2] = {zero, zero};                         // x . W_f (shared by both gates)
    v8f ag0[2] = {zero, zero}, ag1[2] = {zero, zero};  // h_c0 . U_f, h_c1 . U_f

#pragma unroll 1
    for (int k = 0; k < 8; ++k) {  // K = 256 in steps of 32; rolled to cap VGPR liveness
      v16bf aX0 = load_a(sX, 0, k, lane);
      v16bf aX1 = load_a(sX, 1, k, lane);
      {
        v16bf b = load_b(Wip, 48, t, k, lane);
        ai[0] = WMMA_BF16(aX0, b, ai[0]);
        ai[1] = WMMA_BF16(aX1, b, ai[1]);
      }
      {
        v16bf b = load_b(Wip, 48, t + 16, k, lane);
        ao[0] = WMMA_BF16(aX0, b, ao[0]);
        ao[1] = WMMA_BF16(aX1, b, ao[1]);
      }
      {
        v16bf b = load_b(Wip, 48, t + 32, k, lane);
        au[0] = WMMA_BF16(aX0, b, au[0]);
        au[1] = WMMA_BF16(aX1, b, au[1]);
      }
      if (HC) {
        v16bf aH00 = load_a(sH0, 0, k, lane);
        v16bf aH01 = load_a(sH0, 1, k, lane);
        v16bf aH10 = load_a(sH1, 0, k, lane);
        v16bf aH11 = load_a(sH1, 1, k, lane);
        v16bf aHs0 = add_bf(aH00, aH10);   // v_pk_add_bf16, co-executes with WMMA
        v16bf aHs1 = add_bf(aH01, aH11);
        {
          v16bf b = load_b(Uip, 48, t, k, lane);
          ai[0] = WMMA_BF16(aHs0, b, ai[0]);
          ai[1] = WMMA_BF16(aHs1, b, ai[1]);
        }
        {
          v16bf b = load_b(Uip, 48, t + 16, k, lane);
          ao[0] = WMMA_BF16(aHs0, b, ao[0]);
          ao[1] = WMMA_BF16(aHs1, b, ao[1]);
        }
        {
          v16bf b = load_b(Uip, 48, t + 32, k, lane);
          au[0] = WMMA_BF16(aHs0, b, au[0]);
          au[1] = WMMA_BF16(aHs1, b, au[1]);
        }
        {
          v16bf b = load_b(Wfp, 16, t, k, lane);
          afx[0] = WMMA_BF16(aX0, b, afx[0]);
          afx[1] = WMMA_BF16(aX1, b, afx[1]);
        }
        {
          v16bf b = load_b(Ufp, 16, t, k, lane);
          ag0[0] = WMMA_BF16(aH00, b, ag0[0]);
          ag0[1] = WMMA_BF16(aH01, b, ag0[1]);
          ag1[0] = WMMA_BF16(aH10, b, ag1[0]);
          ag1[1] = WMMA_BF16(aH11, b, ag1[1]);
        }
      }
    }

    // ---- in-register epilogue: C/D layout VGPR v -> M = v + 8*(lane>=16), N = lane&15 ----
    const int col = (t << 4) + (lane & 15);
    const int mh  = (lane >> 4) << 3;
    const float bi = b_iou[col];
    const float bo = b_iou[HID + col];
    const float bu = b_iou[2 * HID + col];
    const float bf = HC ? b_f[col] : 0.f;
#pragma unroll
    for (int pt = 0; pt < 2; ++pt) {
#pragma unroll
      for (int v = 0; v < 8; ++v) {
        const int m = (pt << 4) + mh + v;
        const long long node = p0 + m;
        float iv = sigm(ai[pt][v] + bi);
        float ov = sigm(ao[pt][v] + bo);
        float uv = fast_tanh(au[pt][v] + bu);
        float cc = iv * uv;
        if (HC) {
          const long long ch0 = child_base + (long long)(((tile << 5) + m)) * 2;
          float fx = afx[pt][v] + bf;
          float f0 = sigm(fx + ag0[pt][v]);
          float f1 = sigm(fx + ag1[pt][v]);
          cc += f0 * c[ch0 * HID + col] + f1 * c[(ch0 + 1) * HID + col];
        }
        float hh = ov * fast_tanh(cc);
        c[node * HID + col] = cc;
        h[node * HID + col] = (bf16_t)hh;
      }
    }
  }
}

// -------- final classifier: out = sigmoid(h . W_cls + b_cls) --------
__global__ __launch_bounds__(256) void cls_kernel(const bf16_t* __restrict__ h,
                                                  const float* __restrict__ Wc,
                                                  const float* __restrict__ bc,
                                                  float* __restrict__ out, int N) {
  int node = blockIdx.x * blockDim.x + threadIdx.x;
  if (node >= N) return;
  const v8bf* hr = (const v8bf*)(h + (long long)node * HID);
  float acc = 0.f;
#pragma unroll 4
  for (int j = 0; j < HID / 8; ++j) {
    v8bf hv = hr[j];
#pragma unroll
    for (int i = 0; i < 8; ++i) acc += (float)hv[i] * Wc[j * 8 + i];
  }
  out[node] = sigm(acc + bc[0]);
}

extern "C" void kernel_launch(void* const* d_in, const int* in_sizes, int n_in,
                              void* d_out, int out_size, void* d_ws, size_t ws_size,
                              hipStream_t stream) {
  const float* x     = (const float*)d_in[0];
  const float* W_iou = (const float*)d_in[1];
  const float* b_iou = (const float*)d_in[2];
  const float* U_iou = (const float*)d_in[3];
  const float* W_f   = (const float*)d_in[4];
  const float* b_f   = (const float*)d_in[5];
  const float* U_f   = (const float*)d_in[6];
  const float* W_cls = (const float*)d_in[7];
  const float* b_cls = (const float*)d_in[8];

  static const int offs[9] = {0, 65536, 98304, 114688, 122880, 126976, 129024, 130048, 130560};
  const int N = offs[8];

  // workspace layout (~194 MB): h bf16 @0, c f32 @64MB, packed weights @192MB
  char*   ws  = (char*)d_ws;
  bf16_t* h   = (bf16_t*)(ws);
  float*  c   = (float*)(ws + (64ll << 20));
  bf16_t* Wip = (bf16_t*)(ws + (192ll << 20));
  bf16_t* Uip = (bf16_t*)(ws + (192ll << 20) + (1ll << 20));
  bf16_t* Wfp = (bf16_t*)(ws + (192ll << 20) + (2ll << 20));
  bf16_t* Ufp = (bf16_t*)(ws + (192ll << 20) + (3ll << 20));

  pack_weights_kernel<<<(FEAT * 3 * HID + 255) / 256, 256, 0, stream>>>(W_iou, Wip, FEAT, 3 * HID);
  pack_weights_kernel<<<(HID  * 3 * HID + 255) / 256, 256, 0, stream>>>(U_iou, Uip, HID,  3 * HID);
  pack_weights_kernel<<<(FEAT * HID     + 255) / 256, 256, 0, stream>>>(W_f,   Wfp, FEAT, HID);
  pack_weights_kernel<<<(HID  * HID     + 255) / 256, 256, 0, stream>>>(U_f,   Ufp, HID,  HID);

  for (int l = 0; l < 8; ++l) {
    const int base  = offs[l];
    const int lsize = offs[l + 1] - offs[l];
    const int cbase = (l > 0) ? offs[l - 1] : 0;
    if (l == 0)
      level_kernel<false><<<lsize / 32, 256, 0, stream>>>(x, h, c, Wip, Uip, Wfp, Ufp,
                                                          b_iou, b_f, base, cbase);
    else
      level_kernel<true><<<lsize / 32, 256, 0, stream>>>(x, h, c, Wip, Uip, Wfp, Ufp,
                                                         b_iou, b_f, base, cbase);
  }

  cls_kernel<<<(N + 255) / 256, 256, 0, stream>>>(h, W_cls, b_cls, (float*)d_out, N);
}
